// SequentialTransductionUnit_38783554683549
// MI455X (gfx1250) — matrix operand order repaired and checked by reading, not compiled, not run on hardware
//
#include <hip/hip_runtime.h>
#include <math.h>

typedef __attribute__((ext_vector_type(16))) __bf16 v16bf;
typedef __attribute__((ext_vector_type(8)))  float  v8f;
typedef __attribute__((ext_vector_type(4)))  unsigned int u32x4;
typedef __attribute__((ext_vector_type(8)))  int i32x8;
typedef __attribute__((ext_vector_type(4)))  int i32x4;

union FragBF { v16bf v; u32x4 q[2]; };

#define B_  2
#define L_  2048
#define DM  1024
#define NH  16
#define DH  64
#define DP  4096   // 4*DM

__device__ __forceinline__ float silu_f(float x) {
  return x * (1.0f / (1.0f + __expf(-x)));
}

// Low 32 bits of a flat shared-aperture address = LDS byte offset (ISA 10.2).
__device__ __forceinline__ unsigned ldsoff(const void* p) {
  return (unsigned)(uintptr_t)p;
}

// TDM: DMA one 64x64 bf16 tile (row pitch = stride elems) from global into LDS,
// hardware-padding each 128B row by 16B -> LDS row stride 72 bf16 (144B).
// D# group0/group1 per CDNA5 ISA ch.8; groups 2/3 zero (2D tensor).
__device__ __forceinline__ void tdm_tile64(unsigned lds_addr, const __bf16* gptr,
                                           unsigned stride_elems) {
  unsigned long long ga = (unsigned long long)(uintptr_t)gptr;
  u32x4 g0;
  g0[0] = 1u;                                   // count=1, user mode
  g0[1] = lds_addr;                             // lds_addr
  g0[2] = (unsigned)(ga & 0xffffffffu);         // global_addr[31:0]
  g0[3] = (unsigned)((ga >> 32) & 0x01ffffffu)  // global_addr[56:32]
        | (2u << 30);                           // type=2 ("image")
  i32x8 g1;
  g1[0] = (int)((1u << 16)      // data_size=1 -> 2 bytes
        | (1u << 20)            // pad_enable
        | (4u << 22)            // pad_interval: 32 DWORDs (=128B of data)
        | (3u << 25));          // pad_amount: 4 DWORDs (=16B)
  g1[1] = (int)(64u << 16);                     // abar=0 | tensor_dim0=64 lo16
  g1[2] = (int)((64u >> 16) | (64u << 16));     // tensor_dim0 hi | tensor_dim1 lo
  g1[3] = (int)((64u >> 16) | (64u << 16));     // tensor_dim1 hi | tile_dim0=64
  g1[4] = (int)64u;                             // tile_dim1=64 | tile_dim2=0
  g1[5] = (int)stride_elems;                    // tensor_dim0_stride lo32
  g1[6] = 0;                                    // stride hi | dim1_stride lo
  g1[7] = 0;
  i32x4 z4 = {0, 0, 0, 0};
#if __clang_major__ >= 23
  i32x8 z8 = {0, 0, 0, 0, 0, 0, 0, 0};
  __builtin_amdgcn_tensor_load_to_lds(g0, g1, z4, z4, z8, 0);
#else
  __builtin_amdgcn_tensor_load_to_lds(g0, g1, z4, z4, 0);
#endif
}

// A fragment (16x32 bf16): lane holds row (lane&15); K halves {h*8..h*8+7, 16+h*8..}
__device__ __forceinline__ v16bf load_fragA(const __bf16* base, int row0, int stride, int koff) {
  int lane = threadIdx.x & 31;
  int r = lane & 15, hi = lane >> 4;
  const __bf16* p = base + (row0 + r) * stride + koff + hi * 8;
  FragBF f;
  f.q[0] = *(const u32x4*)p;
  f.q[1] = *(const u32x4*)(p + 16);
  return f.v;
}

// B fragment (32x16 bf16): lane holds column (lane&15); K halves {h*8..h*8+7, 16+h*8..}
__device__ __forceinline__ v16bf load_fragB(const __bf16* base, int col0, int stride, int koff) {
  int lane = threadIdx.x & 31;
  int n = lane & 15, hi = lane >> 4;
  const __bf16* p = base + (col0 + n) * stride + koff + hi * 8;
  FragBF f;
  f.q[0] = *(const u32x4*)p;
  f.q[1] = *(const u32x4*)(p + 16);
  return f.v;
}

__global__ void stu_to_bf16(__bf16* __restrict__ dst, const float* __restrict__ src, int n) {
  int i = blockIdx.x * blockDim.x + threadIdx.x;
  if (i < n) dst[i] = (__bf16)src[i];
}

__global__ void stu_cumsum(float* __restrict__ pos, const float* __restrict__ lam) {
  int b = blockIdx.x;
  if (threadIdx.x == 0) {
    float acc = 0.f;
    for (int i = 0; i < L_; ++i) { acc += lam[b * L_ + i]; pos[b * L_ + i] = acc; }
  }
}

// h = silu(x @ W1^T + b), fused RoPE/gates; emits u(fp32), q/k/v (bf16, head-major)
// TDM double-buffered 64-wide k-panels.
__global__ __launch_bounds__(256)
void stu_gemm1_rope(const __bf16* __restrict__ xb, const __bf16* __restrict__ w1,
                    const float* __restrict__ proj_b, const float* __restrict__ pos,
                    const float* __restrict__ q_gate, const float* __restrict__ k_gate,
                    const float* __restrict__ v_gate,
                    float* __restrict__ u_out,
                    __bf16* __restrict__ qh, __bf16* __restrict__ kh, __bf16* __restrict__ vh)
{
  __shared__ __bf16 sA[2][64 * 72];
  __shared__ __bf16 sB[2][64 * 72];
  __shared__ float  stage[64 * 68];

  const int tid = threadIdx.x;
  const int col0 = blockIdx.x * 64;   // over DP=4096
  const int row0 = blockIdx.y * 64;   // over B*L
  const int wave = tid >> 5, lane = tid & 31;
  const int hi = lane >> 4, lc = lane & 15;
  const int t0 = wave * 2;
  const int cm = t0 >> 2, cn0 = t0 & 3, cn1 = cn0 + 1;

  const int nsteps = DM / 64;
  if (wave == 0) {
    tdm_tile64(ldsoff(sA[0]), xb + (size_t)row0 * DM, DM);
    tdm_tile64(ldsoff(sB[0]), w1 + (size_t)col0 * DM, DM);
  }

  v8f acc0 = {}; v8f acc1 = {};

  for (int s = 0; s < nsteps; ++s) {
    __syncthreads();   // consumers done with buffer (s+1)&1 before TDM overwrites it
    if (wave == 0) {
      if (s + 1 < nsteps) {
        tdm_tile64(ldsoff(sA[(s + 1) & 1]), xb + (size_t)row0 * DM + (s + 1) * 64, DM);
        tdm_tile64(ldsoff(sB[(s + 1) & 1]), w1 + (size_t)col0 * DM + (s + 1) * 64, DM);
        __builtin_amdgcn_s_wait_tensorcnt((short)2);  // stage s resident (in-order)
      } else {
        __builtin_amdgcn_s_wait_tensorcnt((short)0);
      }
    }
    __syncthreads();
    const __bf16* cA = sA[s & 1];
    const __bf16* cB = sB[s & 1];
    for (int kk = 0; kk < 64; kk += 32) {
      v16bf a  = load_fragA(cA, cm * 16, 72, kk);
      v16bf b0 = load_fragB(cB, cn0 * 16, 72, kk);
      v16bf b1 = load_fragB(cB, cn1 * 16, 72, kk);
      acc0 = __builtin_amdgcn_wmma_f32_16x16x32_bf16(false, a, false, b0, (short)0, acc0, false, false);
      acc1 = __builtin_amdgcn_wmma_f32_16x16x32_bf16(false, a, false, b1, (short)0, acc1, false, false);
    }
  }

  __syncthreads();
  for (int r = 0; r < 8; ++r) {
    int rl = cm * 16 + r + 8 * hi;
    int c0 = cn0 * 16 + lc, c1 = cn1 * 16 + lc;
    stage[rl * 68 + c0] = silu_f(acc0[r] + proj_b[col0 + c0]);
    stage[rl * 68 + c1] = silu_f(acc1[r] + proj_b[col0 + c1]);
  }
  __syncthreads();

  const int chunk = col0 >> 10;            // 0=u 1=v 2=q 3=k
  const int head  = (col0 & 1023) >> 6;
  for (int idx = tid; idx < 64 * 64; idx += 256) {
    int rl = idx >> 6, cl = idx & 63;
    int rg = row0 + rl;
    int b = rg >> 11, l = rg & 2047;
    if (chunk == 0) {
      u_out[(size_t)rg * DM + head * 64 + cl] = stage[rl * 68 + cl];
    } else if (chunk == 1) {
      float g = v_gate[(b * L_ + l) * NH + head];
      vh[((size_t)(b * NH + head) * L_ + l) * DH + cl] = (__bf16)(stage[rl * 68 + cl] * g);
    } else {
      int i = cl & 31;
      float te = stage[rl * 68 + 2 * i];
      float to = stage[rl * 68 + 2 * i + 1];
      float ang = pos[rg] * __expf(-0.2878231366242557f * (float)i); // 10000^(-2i/64)
      float sn, cs; __sincosf(ang, &sn, &cs);
      float val = (cl < 32) ? (te * cs - to * sn) : (te * sn + to * cs);
      if (chunk == 2) {
        val *= q_gate[(b * L_ + l) * NH + head];
        qh[((size_t)(b * NH + head) * L_ + l) * DH + cl] = (__bf16)val;
      } else {
        val *= k_gate[(b * L_ + l) * NH + head];
        kh[((size_t)(b * NH + head) * L_ + l) * DH + cl] = (__bf16)val;
      }
    }
  }
}

// streaming causal attention, silu-scored (no softmax), fused u-gate on output.
// Q/K tiles arrive via TDM; the V transpose overlaps the DMA.
__global__ __launch_bounds__(256)
void stu_attn(const __bf16* __restrict__ qh, const __bf16* __restrict__ kh,
              const __bf16* __restrict__ vh, const float* __restrict__ bias,
              const float* __restrict__ head_scale, const float* __restrict__ u_in,
              __bf16* __restrict__ a_out)
{
  __shared__ __bf16 sQ[64 * 72];
  __shared__ __bf16 sK[64 * 72];
  __shared__ __bf16 sVT[64 * 72];   // transposed: [d][j]
  __shared__ __bf16 sS[64 * 72];

  const int tid = threadIdx.x;
  const int i0 = blockIdx.x * 64;
  const int bh = blockIdx.y;
  const int b = bh >> 4, head = bh & 15;
  const int wave = tid >> 5, lane = tid & 31;
  const int hi = lane >> 4, lc = lane & 15;
  const int t0 = wave * 2;
  const int cm = t0 >> 2, cn0 = t0 & 3, cn1 = cn0 + 1;

  const float scale = head_scale[head];
  const __bf16* qbase = qh + (size_t)bh * L_ * DH;
  const __bf16* kbase = kh + (size_t)bh * L_ * DH;
  const __bf16* vbase = vh + (size_t)bh * L_ * DH;
  const float*  brow  = bias + (size_t)b * L_ * L_;

  if (wave == 0) tdm_tile64(ldsoff(sQ), qbase + (size_t)i0 * DH, DH);

  v8f y0 = {}; v8f y1 = {};

  for (int j0 = 0; j0 <= i0; j0 += 64) {
    __syncthreads();   // everyone done with sK/sVT/sS of previous block
    if (wave == 0) tdm_tile64(ldsoff(sK), kbase + (size_t)j0 * DH, DH);
    for (int idx = tid; idx < 64 * 64; idx += 256) {   // transpose V into LDS
      int j = idx >> 6, d = idx & 63;
      sVT[d * 72 + j] = vbase[(size_t)(j0 + j) * DH + d];
    }
    if (wave == 0) __builtin_amdgcn_s_wait_tensorcnt((short)0); // sQ (1st iter) + sK
    __syncthreads();

    v8f s0 = {}; v8f s1 = {};
    for (int kk = 0; kk < DH; kk += 32) {
      v16bf a  = load_fragA(sQ, cm * 16, 72, kk);
      v16bf b0 = load_fragB(sK, cn0 * 16, 72, kk);
      v16bf b1 = load_fragB(sK, cn1 * 16, 72, kk);
      s0 = __builtin_amdgcn_wmma_f32_16x16x32_bf16(false, a, false, b0, (short)0, s0, false, false);
      s1 = __builtin_amdgcn_wmma_f32_16x16x32_bf16(false, a, false, b1, (short)0, s1, false, false);
    }
    // scale + 4*bias + silu + causal mask -> bf16 restage
    for (int r = 0; r < 8; ++r) {
      int rl = cm * 16 + r + 8 * hi;
      int ig = i0 + rl;
      int c0 = cn0 * 16 + lc, c1 = cn1 * 16 + lc;
      int jg0 = j0 + c0, jg1 = j0 + c1;
      float v0f = silu_f(s0[r] * scale + 4.0f * brow[(size_t)ig * L_ + jg0]);
      float v1f = silu_f(s1[r] * scale + 4.0f * brow[(size_t)ig * L_ + jg1]);
      if (jg0 > ig) v0f = 0.0f;
      if (jg1 > ig) v1f = 0.0f;
      sS[rl * 72 + c0] = (__bf16)v0f;
      sS[rl * 72 + c1] = (__bf16)v1f;
    }
    __syncthreads();
    for (int kk = 0; kk < 64; kk += 32) {
      v16bf a  = load_fragA(sS, cm * 16, 72, kk);
      v16bf b0 = load_fragB(sVT, cn0 * 16, 72, kk);
      v16bf b1 = load_fragB(sVT, cn1 * 16, 72, kk);
      y0 = __builtin_amdgcn_wmma_f32_16x16x32_bf16(false, a, false, b0, (short)0, y0, false, false);
      y1 = __builtin_amdgcn_wmma_f32_16x16x32_bf16(false, a, false, b1, (short)0, y1, false, false);
    }
  }

  // fuse u-gate; emit bf16 operand for the output GEMM
  for (int r = 0; r < 8; ++r) {
    int rl = cm * 16 + r + 8 * hi;
    int ig = i0 + rl;
    size_t rowg = (size_t)(b * L_ + ig) * DM;
    int d0 = head * 64 + cn0 * 16 + lc;
    int d1 = head * 64 + cn1 * 16 + lc;
    a_out[rowg + d0] = (__bf16)(y0[r] * u_in[rowg + d0]);
    a_out[rowg + d1] = (__bf16)(y1[r] * u_in[rowg + d1]);
  }
}

// r = x + (a @ W2^T + out_b); TDM double-buffered panels
__global__ __launch_bounds__(256)
void stu_gemm2_res(const __bf16* __restrict__ ab, const __bf16* __restrict__ w2,
                   const float* __restrict__ out_b, const float* __restrict__ x,
                   float* __restrict__ r_out)
{
  __shared__ __bf16 sA[2][64 * 72];
  __shared__ __bf16 sB[2][64 * 72];
  const int tid = threadIdx.x;
  const int col0 = blockIdx.x * 64;
  const int row0 = blockIdx.y * 64;
  const int wave = tid >> 5, lane = tid & 31;
  const int hi = lane >> 4, lc = lane & 15;
  const int t0 = wave * 2;
  const int cm = t0 >> 2, cn0 = t0 & 3, cn1 = cn0 + 1;

  const int nsteps = DM / 64;
  if (wave == 0) {
    tdm_tile64(ldsoff(sA[0]), ab + (size_t)row0 * DM, DM);
    tdm_tile64(ldsoff(sB[0]), w2 + (size_t)col0 * DM, DM);
  }

  v8f acc0 = {}; v8f acc1 = {};
  for (int s = 0; s < nsteps; ++s) {
    __syncthreads();
    if (wave == 0) {
      if (s + 1 < nsteps) {
        tdm_tile64(ldsoff(sA[(s + 1) & 1]), ab + (size_t)row0 * DM + (s + 1) * 64, DM);
        tdm_tile64(ldsoff(sB[(s + 1) & 1]), w2 + (size_t)col0 * DM + (s + 1) * 64, DM);
        __builtin_amdgcn_s_wait_tensorcnt((short)2);
      } else {
        __builtin_amdgcn_s_wait_tensorcnt((short)0);
      }
    }
    __syncthreads();
    const __bf16* cA = sA[s & 1];
    const __bf16* cB = sB[s & 1];
    for (int kk = 0; kk < 64; kk += 32) {
      v16bf a  = load_fragA(cA, cm * 16, 72, kk);
      v16bf b0 = load_fragB(cB, cn0 * 16, 72, kk);
      v16bf b1 = load_fragB(cB, cn1 * 16, 72, kk);
      acc0 = __builtin_amdgcn_wmma_f32_16x16x32_bf16(false, a, false, b0, (short)0, acc0, false, false);
      acc1 = __builtin_amdgcn_wmma_f32_16x16x32_bf16(false, a, false, b1, (short)0, acc1, false, false);
    }
  }
  for (int r = 0; r < 8; ++r) {
    int rl = cm * 16 + r + 8 * hi;
    size_t rg = (size_t)(row0 + rl);
    int c0 = col0 + cn0 * 16 + lc;
    int c1 = col0 + cn1 * 16 + lc;
    r_out[rg * DM + c0] = x[rg * DM + c0] + acc0[r] + out_b[c0];
    r_out[rg * DM + c1] = x[rg * DM + c1] + acc1[r] + out_b[c1];
  }
}

__global__ __launch_bounds__(256)
void stu_layernorm(const float* __restrict__ r, const float* __restrict__ gamma,
                   const float* __restrict__ beta, float* __restrict__ out)
{
  __shared__ float red[256];
  const int row = blockIdx.x;
  const float* rr = r + (size_t)row * DM;
  float vals[4];
  float s = 0.f, sq = 0.f;
  for (int i = 0; i < 4; ++i) {
    float v = rr[threadIdx.x + i * 256];
    vals[i] = v; s += v; sq += v * v;
  }
  red[threadIdx.x] = s; __syncthreads();
  for (int off = 128; off > 0; off >>= 1) {
    if (threadIdx.x < off) red[threadIdx.x] += red[threadIdx.x + off];
    __syncthreads();
  }
  float mean = red[0] * (1.0f / DM);
  __syncthreads();
  red[threadIdx.x] = sq; __syncthreads();
  for (int off = 128; off > 0; off >>= 1) {
    if (threadIdx.x < off) red[threadIdx.x] += red[threadIdx.x + off];
    __syncthreads();
  }
  float var = red[0] * (1.0f / DM) - mean * mean;
  float inv = rsqrtf(var + 1e-5f);
  for (int i = 0; i < 4; ++i) {
    int c = threadIdx.x + i * 256;
    out[(size_t)row * DM + c] = (vals[i] - mean) * inv * gamma[c] + beta[c];
  }
}

extern "C" void kernel_launch(void* const* d_in, const int* in_sizes, int n_in,
                              void* d_out, int out_size, void* d_ws, size_t ws_size,
                              hipStream_t stream) {
  (void)in_sizes; (void)n_in; (void)out_size; (void)ws_size;
  const float* x           = (const float*)d_in[0];
  // d_in[1] attn_mask: causal tril, applied analytically. d_in[2] ts: unused by reference.
  const float* rope_lambda = (const float*)d_in[3];
  const float* q_gate      = (const float*)d_in[4];
  const float* k_gate      = (const float*)d_in[5];
  const float* v_gate      = (const float*)d_in[6];
  const float* tpb         = (const float*)d_in[7];
  const float* proj_w      = (const float*)d_in[8];
  const float* proj_b      = (const float*)d_in[9];
  const float* out_w       = (const float*)d_in[10];
  const float* out_b       = (const float*)d_in[11];
  const float* head_scale  = (const float*)d_in[12];
  const float* ln_g        = (const float*)d_in[13];
  const float* ln_b        = (const float*)d_in[14];
  float* out = (float*)d_out;

  char* ws = (char*)d_ws;
  size_t off = 0;
  auto alloc = [&](size_t bytes) -> char* {
    char* p = ws + off; off += (bytes + 255) & ~(size_t)255; return p;
  };
  __bf16* xb   = (__bf16*)alloc((size_t)B_ * L_ * DM * 2);
  __bf16* w1b  = (__bf16*)alloc((size_t)DP * DM * 2);
  __bf16* w2b  = (__bf16*)alloc((size_t)DM * DM * 2);
  float*  pos  = (float*) alloc((size_t)B_ * L_ * 4);
  float*  u    = (float*) alloc((size_t)B_ * L_ * DM * 4);
  __bf16* qhp  = (__bf16*)alloc((size_t)B_ * NH * L_ * DH * 2);
  __bf16* khp  = (__bf16*)alloc((size_t)B_ * NH * L_ * DH * 2);
  __bf16* vhp  = (__bf16*)alloc((size_t)B_ * NH * L_ * DH * 2);
  __bf16* abuf = (__bf16*)alloc((size_t)B_ * L_ * DM * 2);
  float*  rres = (float*) alloc((size_t)B_ * L_ * DM * 4);

  int n1 = B_ * L_ * DM;
  stu_to_bf16<<<(n1 + 255) / 256, 256, 0, stream>>>(xb, x, n1);
  int n2 = DP * DM;
  stu_to_bf16<<<(n2 + 255) / 256, 256, 0, stream>>>(w1b, proj_w, n2);
  int n3 = DM * DM;
  stu_to_bf16<<<(n3 + 255) / 256, 256, 0, stream>>>(w2b, out_w, n3);
  stu_cumsum<<<B_, 32, 0, stream>>>(pos, rope_lambda);

  stu_gemm1_rope<<<dim3(DP / 64, (B_ * L_) / 64), 256, 0, stream>>>(
      xb, w1b, proj_b, pos, q_gate, k_gate, v_gate, u, qhp, khp, vhp);

  stu_attn<<<dim3(L_ / 64, B_ * NH), 256, 0, stream>>>(
      qhp, khp, vhp, tpb, head_scale, u, abuf);

  stu_gemm2_res<<<dim3(DM / 64, (B_ * L_) / 64), 256, 0, stream>>>(
      abuf, w2b, out_b, x, rres);

  stu_layernorm<<<B_ * L_, 256, 0, stream>>>(rres, ln_g, ln_b, out);
}
